// GraphRecModel_37529424233117
// MI455X (gfx1250) — matrix operand order repaired and checked by reading, not compiled, not run on hardware
//
#include <hip/hip_runtime.h>
#include <hip/hip_bf16.h>

typedef __bf16 bf16_t;
typedef __attribute__((ext_vector_type(16))) __bf16 v16bf;
typedef __attribute__((ext_vector_type(8)))  float  v8f;

#define EPSV 1e-5f

// Load 8 contiguous floats (two b128 loads).
__device__ __forceinline__ void load8(const float* __restrict__ p, float* v) {
    float4 a = ((const float4*)p)[0];
    float4 b = ((const float4*)p)[1];
    v[0]=a.x; v[1]=a.y; v[2]=a.z; v[3]=a.w;
    v[4]=b.x; v[5]=b.y; v[6]=b.z; v[7]=b.w;
}

// ------------------------------------------------------------------
// 64-row-strip WMMA GEMM:  Y[M,NOUT] (+)= act(A')[M,64] @ W[64,NOUT] (+ bias)
//   GATHER: A'(row,k) = xu[gs[row]][k] * xv[gd[row]][k]   (fused u_mul_v)
//   AFFINE: a = relu(a*asc[k] + ash[k])                    (fused BN+ReLU)
//   ACC:    Y += (HeteroGraphConv 'sum' aggregation)
//   STATS:  atomic column sum/sumsq side-output (BN statistics)
// One wave per block; 4 row-tiles of 16 share each B fragment.
// Wave-uniform `full` flag: all-but-last blocks take a branch-free path.
// A-fragment K layout (ISA 7.12.2, 16-bit A 16x32): per lane-half the 16
// elements are two contiguous 8-float runs -> b128 vector loads.
// ------------------------------------------------------------------
template<bool GATHER, bool AFFINE, bool ACC, bool STATS, bool BIAS, int NOUT>
__global__ void __launch_bounds__(32)
gr_gemm(const float* __restrict__ A,
        const int* __restrict__ gs, const int* __restrict__ gd,
        const float* __restrict__ xu, const float* __restrict__ xv,
        const float* __restrict__ asc, const float* __restrict__ ash,
        const float* __restrict__ W, const float* __restrict__ bias,
        float* __restrict__ Y, int M,
        float* __restrict__ stats)
{
    const int  l    = threadIdx.x;
    const int  half = l >> 4;
    const int  lm   = l & 15;
    const long row0 = (long)blockIdx.x * 64;
    const bool full = (row0 + 64) <= (long)M;      // wave-uniform

    // ---- A fragments: 4 row tiles x 2 K-blocks ----
    v16bf af[4][2];
#pragma unroll
    for (int rt = 0; rt < 4; ++rt) {
        const long arow = row0 + rt * 16 + lm;
        const bool rok  = full || (arow < (long)M);
        const float* pa;
        const float* pb = nullptr;
        if (GATHER) {
            int s = 0, d = 0;
            if (full)      { s = gs[arow]; d = gd[arow]; }
            else if (rok)  { s = gs[arow]; d = gd[arow]; }
            pa = xu + (long)s * 64;
            pb = xv + (long)d * 64;
        } else {
            pa = A + arow * 64;
        }
#pragma unroll
        for (int kb = 0; kb < 2; ++kb) {
            const int k0 = kb * 32 + 8 * half;        // elements 0..7
            const int k1 = k0 + 16;                   // elements 8..15
            float va[16];
            if (rok) {
                load8(pa + k0, va);
                load8(pa + k1, va + 8);
                if (GATHER) {
                    float vb[16];
                    load8(pb + k0, vb);
                    load8(pb + k1, vb + 8);
#pragma unroll
                    for (int i = 0; i < 16; ++i) va[i] *= vb[i];
                }
                if (AFFINE) {
                    float sv[16], hv[16];
                    load8(asc + k0, sv); load8(asc + k1, sv + 8);
                    load8(ash + k0, hv); load8(ash + k1, hv + 8);
#pragma unroll
                    for (int i = 0; i < 16; ++i)
                        va[i] = fmaxf(va[i] * sv[i] + hv[i], 0.f);
                }
            } else {
#pragma unroll
                for (int i = 0; i < 16; ++i) va[i] = 0.f;
            }
#pragma unroll
            for (int e = 0; e < 16; ++e) af[rt][kb][e] = (bf16_t)va[e];
        }
    }

    // ---- N tiles (fully unrolled; NOUT is compile-time) ----
    constexpr int NTILES = NOUT >> 4;
#pragma unroll
    for (int nt = 0; nt < NTILES; ++nt) {
        const int n = nt * 16 + lm;
        v16bf bf0, bf1;                               // B 32x16: col-per-lane
#pragma unroll
        for (int e = 0; e < 16; ++e) {
            bf0[e] = (bf16_t)W[(8 * half + e + (e >= 8 ? 8 : 0)) * NOUT + n];
            bf1[e] = (bf16_t)W[(32 + 8 * half + e + (e >= 8 ? 8 : 0)) * NOUT + n];
        }
        const float bv = BIAS ? bias[n] : 0.f;
        float ssum = 0.f, ssq = 0.f;
#pragma unroll
        for (int rt = 0; rt < 4; ++rt) {
            v8f c = {};
            c = __builtin_amdgcn_wmma_f32_16x16x32_bf16(
                    false, af[rt][0], false, bf0, (short)0, c, false, false);
            c = __builtin_amdgcn_wmma_f32_16x16x32_bf16(
                    false, af[rt][1], false, bf1, (short)0, c, false, false);
            // C/D layout: lane holds rows r + 8*half, column n
            float* yp = Y + (row0 + rt * 16 + 8 * half) * (long)NOUT + n;
            if (full) {                                // branch-free hot path
#pragma unroll
                for (int r = 0; r < 8; ++r) {
                    float y = c[r] + bv;
                    if (ACC) y += yp[r * NOUT];
                    yp[r * NOUT] = y;
                    if (STATS) { ssum += y; ssq += y * y; }
                }
            } else {                                   // tail block only
#pragma unroll
                for (int r = 0; r < 8; ++r) {
                    long orow = row0 + rt * 16 + r + 8 * half;
                    if (orow < (long)M) {
                        float y = c[r] + bv;
                        if (ACC) y += yp[r * NOUT];
                        yp[r * NOUT] = y;
                        if (STATS) { ssum += y; ssq += y * y; }
                    }
                }
            }
        }
        if (STATS) {
            atomicAdd(&stats[n], ssum);
            atomicAdd(&stats[NOUT + n], ssq);
        }
    }
}

// ---- zero a float region (grid-stride) ----
__global__ void gr_zero(float* __restrict__ p, long n) {
    long i = (long)blockIdx.x * blockDim.x + threadIdx.x;
    long s = (long)gridDim.x * blockDim.x;
    for (; i < n; i += s) p[i] = 0.f;
}

// ---- scatter-add of 64-wide features + degree count ----
__global__ void gr_scatter(const float* __restrict__ feat, const int* __restrict__ src,
                           const int* __restrict__ dst, float* __restrict__ sum,
                           float* __restrict__ deg, int E)
{
    long t = (long)blockIdx.x * blockDim.x + threadIdx.x;
    int  e = (int)(t >> 6);
    int  f = (int)(t & 63);
    if (e < E) {
        int s = src[e], d = dst[e];
        atomicAdd(&sum[(long)d * 64 + f], feat[(long)s * 64 + f]);
        if (f == 0) atomicAdd(&deg[d], 1.0f);
    }
}

// ---- sum / max(deg,1) in place ----
__global__ void gr_divdeg(float* __restrict__ sum, const float* __restrict__ deg, int N) {
    long t = (long)blockIdx.x * blockDim.x + threadIdx.x;
    int  i = (int)(t >> 6);
    int  f = (int)(t & 63);
    if (i < N) {
        float d = deg[i];
        d = d > 1.f ? d : 1.f;
        sum[(long)i * 64 + f] /= d;
    }
}

// ---- BN finalize: stats(sum,sumsq) -> per-column scale/shift ----
__global__ void gr_bnfin(const float* __restrict__ stats, const float* __restrict__ gamma,
                         const float* __restrict__ beta, float* __restrict__ sc,
                         float* __restrict__ sh, int C, float invM)
{
    int c = threadIdx.x;
    if (c < C) {
        float mean = stats[c] * invM;
        float var  = stats[C + c] * invM - mean * mean;
        float inv  = rsqrtf(var + EPSV);
        float s    = gamma[c] * inv;
        sc[c] = s;
        sh[c] = beta[c] - mean * s;
    }
}

// ---- final: relu(bn4(t2)) @ Wuv3 + buv3 ----
__global__ void gr_score(const float* __restrict__ t2, const float* __restrict__ sc,
                         const float* __restrict__ sh, const float* __restrict__ w3,
                         const float* __restrict__ b3, float* __restrict__ out, int E)
{
    long e = (long)blockIdx.x * blockDim.x + threadIdx.x;
    if (e < (long)E) {
        float v[16];
        load8(t2 + e * 16, v);
        load8(t2 + e * 16 + 8, v + 8);
        float a = b3[0];
#pragma unroll
        for (int c = 0; c < 16; ++c) {
            float x = fmaxf(v[c] * sc[c] + sh[c], 0.f);
            a += x * w3[c];
        }
        out[e] = a;
    }
}

extern "C" void kernel_launch(void* const* d_in, const int* in_sizes, int n_in,
                              void* d_out, int out_size, void* d_ws, size_t ws_size,
                              hipStream_t stream)
{
    (void)n_in; (void)ws_size; (void)out_size;
    const float* u2e     = (const float*)d_in[0];
    const float* v2e     = (const float*)d_in[1];
    const float* W_self  = (const float*)d_in[2];   // [3,64,64]
    const float* b_self  = (const float*)d_in[3];   // [3,64]
    const float* W_neigh = (const float*)d_in[4];   // [3,64,64]
    const float* Wur1 = (const float*)d_in[5];  const float* bur1 = (const float*)d_in[6];
    const float* Wur2 = (const float*)d_in[7];  const float* bur2 = (const float*)d_in[8];
    const float* Wvr1 = (const float*)d_in[9];  const float* bvr1 = (const float*)d_in[10];
    const float* Wvr2 = (const float*)d_in[11]; const float* bvr2 = (const float*)d_in[12];
    const float* Wuv1 = (const float*)d_in[13]; const float* buv1 = (const float*)d_in[14];
    const float* Wuv2 = (const float*)d_in[15]; const float* buv2 = (const float*)d_in[16];
    const float* Wuv3 = (const float*)d_in[17]; const float* buv3 = (const float*)d_in[18];
    const float* bn_g  = (const float*)d_in[19];    // [3,64]
    const float* bn_b  = (const float*)d_in[20];
    const float* bn4_g = (const float*)d_in[21];    // [16]
    const float* bn4_b = (const float*)d_in[22];
    const int* social_src = (const int*)d_in[23];
    const int* social_dst = (const int*)d_in[24];
    const int* rates_src  = (const int*)d_in[25];
    const int* rates_dst  = (const int*)d_in[26];

    const int NU = in_sizes[0] / 64;
    const int NV = in_sizes[1] / 64;
    const int ES = in_sizes[23];
    const int ER = in_sizes[25];

    float* w = (float*)d_ws;
    size_t off = 0;
    auto alloc = [&](size_t n) { float* p = w + off; off += n; return p; };

    // accumulated region (zeroed every call) — contiguous
    float* sumS = alloc((size_t)NU * 64);  float* degS = alloc(NU);
    float* sumR = alloc((size_t)NU * 64);  float* degR = alloc(NU);
    float* sumI = alloc((size_t)NV * 64);  float* degI = alloc(NV);
    float* st_tu = alloc(128); float* st_tv = alloc(128);
    float* st_t1 = alloc(128); float* st_t2 = alloc(32);
    const size_t nz = off;
    // non-accumulated
    float* sc_tu = alloc(64); float* sh_tu = alloc(64);
    float* sc_tv = alloc(64); float* sh_tv = alloc(64);
    float* sc_t1 = alloc(64); float* sh_t1 = alloc(64);
    float* sc_t2 = alloc(16); float* sh_t2 = alloc(16);
    float* h_u = alloc((size_t)NU * 64);
    float* h_v = alloc((size_t)NV * 64);
    float* t_u = alloc((size_t)NU * 64);
    float* t_v = alloc((size_t)NV * 64);
    float* x_u = alloc((size_t)NU * 64);
    float* x_v = alloc((size_t)NV * 64);
    float* t1  = alloc((size_t)ER * 64);
    float* t2  = alloc((size_t)ER * 16);

    // 1) zero accumulators (inside the call: safe for graph replay)
    gr_zero<<<4096, 256, 0, stream>>>(w, (long)nz);

    // 2) segment sums + degrees, then divide -> neighbor means (in place)
    {
        dim3 b(256);
        gr_scatter<<<dim3(((size_t)ES * 64 + 255) / 256), b, 0, stream>>>(u2e, social_src, social_dst, sumS, degS, ES);
        gr_scatter<<<dim3(((size_t)ER * 64 + 255) / 256), b, 0, stream>>>(v2e, rates_dst,  rates_src,  sumR, degR, ER);
        gr_scatter<<<dim3(((size_t)ER * 64 + 255) / 256), b, 0, stream>>>(u2e, rates_src,  rates_dst,  sumI, degI, ER);
        gr_divdeg<<<dim3(((size_t)NU * 64 + 255) / 256), b, 0, stream>>>(sumS, degS, NU);
        gr_divdeg<<<dim3(((size_t)NU * 64 + 255) / 256), b, 0, stream>>>(sumR, degR, NU);
        gr_divdeg<<<dim3(((size_t)NV * 64 + 255) / 256), b, 0, stream>>>(sumI, degI, NV);
    }

    const dim3 wb(32);
    const dim3 gU((NU + 63) / 64), gV((NV + 63) / 64), gE((ER + 63) / 64);

    // 3) h_u = u2e@Ws0+bs0 + nS@Wn0 + u2e@Ws2+bs2 + nR@Wn2   (HeteroConv sum)
    gr_gemm<false,false,false,false,true ,64><<<gU, wb, 0, stream>>>(u2e,  nullptr,nullptr,nullptr,nullptr, nullptr,nullptr,
                                        W_self + 0*4096, b_self + 0*64, h_u, NU, nullptr);
    gr_gemm<false,false,true ,false,false,64><<<gU, wb, 0, stream>>>(sumS, nullptr,nullptr,nullptr,nullptr, nullptr,nullptr,
                                        W_neigh + 0*4096, nullptr,      h_u, NU, nullptr);
    gr_gemm<false,false,true ,false,true ,64><<<gU, wb, 0, stream>>>(u2e,  nullptr,nullptr,nullptr,nullptr, nullptr,nullptr,
                                        W_self + 2*4096, b_self + 2*64, h_u, NU, nullptr);
    gr_gemm<false,false,true ,false,false,64><<<gU, wb, 0, stream>>>(sumR, nullptr,nullptr,nullptr,nullptr, nullptr,nullptr,
                                        W_neigh + 2*4096, nullptr,      h_u, NU, nullptr);
    // h_v = v2e@Ws1+bs1 + nI@Wn1
    gr_gemm<false,false,false,false,true ,64><<<gV, wb, 0, stream>>>(v2e,  nullptr,nullptr,nullptr,nullptr, nullptr,nullptr,
                                        W_self + 1*4096, b_self + 1*64, h_v, NV, nullptr);
    gr_gemm<false,false,true ,false,false,64><<<gV, wb, 0, stream>>>(sumI, nullptr,nullptr,nullptr,nullptr, nullptr,nullptr,
                                        W_neigh + 1*4096, nullptr,      h_v, NV, nullptr);

    // 4) user tower
    gr_gemm<false,false,false,true ,true ,64><<<gU, wb, 0, stream>>>(h_u, nullptr,nullptr,nullptr,nullptr, nullptr,nullptr,
                                        Wur1, bur1, t_u, NU, st_tu);
    gr_bnfin<<<1, 64, 0, stream>>>(st_tu, bn_g + 0*64, bn_b + 0*64, sc_tu, sh_tu, 64, 1.0f / NU);
    gr_gemm<false,true ,false,false,true ,64><<<gU, wb, 0, stream>>>(t_u, nullptr,nullptr,nullptr,nullptr, sc_tu, sh_tu,
                                        Wur2, bur2, x_u, NU, nullptr);

    // 5) item tower
    gr_gemm<false,false,false,true ,true ,64><<<gV, wb, 0, stream>>>(h_v, nullptr,nullptr,nullptr,nullptr, nullptr,nullptr,
                                        Wvr1, bvr1, t_v, NV, st_tv);
    gr_bnfin<<<1, 64, 0, stream>>>(st_tv, bn_g + 1*64, bn_b + 1*64, sc_tv, sh_tv, 64, 1.0f / NV);
    gr_gemm<false,true ,false,false,true ,64><<<gV, wb, 0, stream>>>(t_v, nullptr,nullptr,nullptr,nullptr, sc_tv, sh_tv,
                                        Wvr2, bvr2, x_v, NV, nullptr);

    // 6) edge MLP: t1 = (x_u[src] * x_v[dst]) @ Wuv1 + buv1  (gather-mul fused)
    gr_gemm<true ,false,false,true ,true ,64><<<gE, wb, 0, stream>>>(nullptr, rates_src, rates_dst, x_u, x_v, nullptr,nullptr,
                                        Wuv1, buv1, t1, ER, st_t1);
    gr_bnfin<<<1, 64, 0, stream>>>(st_t1, bn_g + 2*64, bn_b + 2*64, sc_t1, sh_t1, 64, 1.0f / ER);
    // t2 = relu(bn(t1)) @ Wuv2 + buv2   (NOUT = 16)
    gr_gemm<false,true ,false,true ,true ,16><<<gE, wb, 0, stream>>>(t1, nullptr,nullptr,nullptr,nullptr, sc_t1, sh_t1,
                                        Wuv2, buv2, t2, ER, st_t2);
    gr_bnfin<<<1, 16, 0, stream>>>(st_t2, bn4_g, bn4_b, sc_t2, sh_t2, 16, 1.0f / ER);

    // 7) scores = relu(bn4(t2)) @ Wuv3 + buv3
    gr_score<<<dim3((ER + 255) / 256), dim3(256), 0, stream>>>(t2, sc_t2, sh_t2, Wuv3, buv3, (float*)d_out, ER);
}
